// GATGraphClassifier_3779571220498
// MI455X (gfx1250) — matrix operand order repaired and checked by reading, not compiled, not run on hardware
//
#include <hip/hip_runtime.h>

// ---------------------------------------------------------------------------
// GAT graph classifier for MI455X (gfx1250, wave32, WMMA).
//   layer GEMMs  : split-bf16 V_WMMA_F32_16X16X32_BF16 (hi*hi + hi*lo + lo*hi)
//   edge softmax : 3-pass (atomic max, exp + atomic sum, normalized scatter)
//   pooling/head : atomics + tiny per-graph log-softmax
// ---------------------------------------------------------------------------

typedef __attribute__((ext_vector_type(16))) __bf16 v16bf;
typedef __attribute__((ext_vector_type(8)))  float  v8f;

// ---------------- fill ----------------
__global__ void fill_kernel(float* __restrict__ p, size_t n, float v) {
    size_t i = (size_t)blockIdx.x * blockDim.x + threadIdx.x;
    if (i < n) p[i] = v;
}

// ---------------- WMMA GEMM: C[nrows,COLS] = A[nrows,K] @ W[K,COLS] ----------------
// One wave computes a 16-row x COLS tile. fp32 emulated via bf16 hi/lo split.
template <int K, int COLS>
__global__ __launch_bounds__(256) void gemm_wmma_splitbf16(
    const float* __restrict__ A, const float* __restrict__ W,
    float* __restrict__ C, int nrows)
{
    const int wave    = threadIdx.x >> 5;
    const int lane    = threadIdx.x & 31;
    const int rowBase = (blockIdx.x * 8 + wave) * 16;
    if (rowBase >= nrows) return;

    const int half = lane >> 4;   // which 16-lane half of the wave
    const int l15  = lane & 15;

    constexpr int NCT = COLS / 16;
    v8f acc[NCT];
#pragma unroll
    for (int ct = 0; ct < NCT; ++ct) acc[ct] = (v8f)(0.0f);

    const int arow = rowBase + l15;

    for (int k0 = 0; k0 < K; k0 += 32) {
        // ---- A fragment: 16-bit A 16x32 layout (ISA 7.12.2):
        //   lane half h, element e: K = 8*h + (e&7) + ((e>>3)*16)
        const float* ap = A + (size_t)arow * K + k0 + half * 8;
        const float4 q0 = *(const float4*)(ap + 0);    // K = base+0..3
        const float4 q1 = *(const float4*)(ap + 4);    // K = base+4..7
        const float4 q2 = *(const float4*)(ap + 16);   // K = base+16..19
        const float4 q3 = *(const float4*)(ap + 20);   // K = base+20..23
        const float af[16] = { q0.x,q0.y,q0.z,q0.w, q1.x,q1.y,q1.z,q1.w,
                               q2.x,q2.y,q2.z,q2.w, q3.x,q3.y,q3.z,q3.w };
        v16bf a_hi, a_lo;
#pragma unroll
        for (int e = 0; e < 16; ++e) {
            float f  = af[e];
            __bf16 h = (__bf16)f;
            a_hi[e]  = h;
            a_lo[e]  = (__bf16)(f - (float)h);
        }

#pragma unroll
        for (int ct = 0; ct < NCT; ++ct) {
            // ---- B fragment: 16-bit B 32x16: lane = column, element e: K = 8*h + (e&7) + ((e>>3)*16)
            const float* wp = W + (size_t)(k0 + half * 8) * COLS + ct * 16 + l15;
            v16bf b_hi, b_lo;
#pragma unroll
            for (int e = 0; e < 16; ++e) {
                float f  = wp[(size_t)((e & 7) + ((e >> 3) << 4)) * COLS];
                __bf16 h = (__bf16)f;
                b_hi[e]  = h;
                b_lo[e]  = (__bf16)(f - (float)h);
            }
            // fp32 ~= hi*hi + hi*lo + lo*hi   (lo*lo ~ 2^-32, dropped)
            acc[ct] = __builtin_amdgcn_wmma_f32_16x16x32_bf16(
                false, a_hi, false, b_hi, (short)0, acc[ct], false, false);
            acc[ct] = __builtin_amdgcn_wmma_f32_16x16x32_bf16(
                false, a_hi, false, b_lo, (short)0, acc[ct], false, false);
            acc[ct] = __builtin_amdgcn_wmma_f32_16x16x32_bf16(
                false, a_lo, false, b_hi, (short)0, acc[ct], false, false);
        }
    }

    // ---- C writeback: VGPR r, lane l -> M = r + 8*(l>>4), N = l&15
#pragma unroll
    for (int ct = 0; ct < NCT; ++ct) {
#pragma unroll
        for (int r = 0; r < 8; ++r) {
            int row = rowBase + r + half * 8;
            C[(size_t)row * COLS + ct * 16 + l15] = acc[ct][r];
        }
    }
}

// ---------------- per-node attention coefficients ----------------
// as[n,h] = sum_c h[n,h,c]*att_s[h,c] ; ad likewise
__global__ void att_coef_kernel(const float* __restrict__ h,
                                const float* __restrict__ att_s,
                                const float* __restrict__ att_d,
                                float* __restrict__ as, float* __restrict__ ad,
                                int n, int heads, int ch)
{
    int t = blockIdx.x * blockDim.x + threadIdx.x;
    if (t >= n * heads) return;
    int node = t / heads, hd = t % heads;
    const float* hp = h + (size_t)node * heads * ch + (size_t)hd * ch;
    const float* sp = att_s + hd * ch;
    const float* dp = att_d + hd * ch;
    float ss = 0.f, dd = 0.f;
    for (int c = 0; c < ch; ++c) { float v = hp[c]; ss += v * sp[c]; dd += v * dp[c]; }
    as[t] = ss; ad[t] = dd;
}

// ---------------- float atomic max via int/uint atomics ----------------
__device__ inline void atomicMaxFloat(float* addr, float val) {
    if (val >= 0.f) atomicMax((int*)addr, __float_as_int(val));
    else            atomicMin((unsigned int*)addr, (unsigned int)__float_as_int(val));
}

__device__ inline void edge_endpoints(const int* __restrict__ ei, int E, int e,
                                      int& src, int& dst) {
    if (e < E) { src = ei[e]; dst = ei[E + e]; }   // edge_index rows 0/1
    else       { src = dst = e - E; }              // self loop
}

// ---------------- pass 1: leaky-relu score + segment max ----------------
__global__ void edge_max_kernel(const int* __restrict__ ei, int E, int Etot,
                                const float* __restrict__ as, const float* __restrict__ ad,
                                float* __restrict__ scores, float* __restrict__ m, int heads)
{
    int e = blockIdx.x * blockDim.x + threadIdx.x;
    if (e >= Etot) return;
    int src, dst; edge_endpoints(ei, E, e, src, dst);
    for (int h = 0; h < heads; ++h) {
        float sc = as[(size_t)src * heads + h] + ad[(size_t)dst * heads + h];
        sc = sc > 0.f ? sc : 0.2f * sc;                         // LeakyReLU(0.2)
        scores[(size_t)e * heads + h] = sc;
        atomicMaxFloat(&m[(size_t)dst * heads + h], sc);
    }
}

// ---------------- pass 2: w = exp(s - m[dst]); den[dst] += w ----------------
__global__ void edge_exp_kernel(const int* __restrict__ ei, int E, int Etot,
                                float* __restrict__ scores, const float* __restrict__ m,
                                float* __restrict__ den, int heads)
{
    int e = blockIdx.x * blockDim.x + threadIdx.x;
    if (e >= Etot) return;
    int src, dst; edge_endpoints(ei, E, e, src, dst);
    for (int h = 0; h < heads; ++h) {
        size_t si = (size_t)e * heads + h;
        float w = expf(scores[si] - m[(size_t)dst * heads + h]);
        scores[si] = w;                                          // overwrite with weight
        atomicAdd(&den[(size_t)dst * heads + h], w);
    }
}

// ---------------- pass 3: out[dst,c] += (w/den[dst]) * h[src,c] ----------------
// C = 1<<cshift channels, head = c >> hshift
__global__ void edge_scatter_kernel(const int* __restrict__ ei, int E, int Etot,
                                    const float* __restrict__ w, const float* __restrict__ den,
                                    const float* __restrict__ h, float* __restrict__ out,
                                    int cshift, int hshift, int heads)
{
    size_t t = (size_t)blockIdx.x * blockDim.x + threadIdx.x;
    size_t total = (size_t)Etot << cshift;
    if (t >= total) return;
    int e = (int)(t >> cshift);
    int c = (int)(t & ((1u << cshift) - 1u));
    int src, dst; edge_endpoints(ei, E, e, src, dst);
    int hd = c >> hshift;
    float alpha = w[(size_t)e * heads + hd] / den[(size_t)dst * heads + hd];
    atomicAdd(&out[((size_t)dst << cshift) + c],
              alpha * h[((size_t)src << cshift) + c]);
}

// ---------------- bias + ELU in place ----------------
__global__ void bias_elu_kernel(float* __restrict__ x, const float* __restrict__ b,
                                size_t total, int cmask)
{
    size_t t = (size_t)blockIdx.x * blockDim.x + threadIdx.x;
    if (t >= total) return;
    float v = x[t] + b[t & (size_t)cmask];
    x[t] = v > 0.f ? v : expm1f(v);                              // ELU(alpha=1)
}

// ---------------- global mean pool: accumulate sums + counts ----------------
__global__ void pool_kernel(const float* __restrict__ h, const int* __restrict__ batch,
                            float* __restrict__ psum, float* __restrict__ pcnt,
                            int n, int cshift)
{
    size_t t = (size_t)blockIdx.x * blockDim.x + threadIdx.x;
    size_t total = (size_t)n << cshift;
    if (t >= total) return;
    int node = (int)(t >> cshift);
    int c    = (int)(t & ((1u << cshift) - 1u));
    int g = batch[node];
    atomicAdd(&psum[((size_t)g << cshift) + c], h[t]);
    if (c == 0) atomicAdd(&pcnt[g], 1.0f);
}

// ---------------- final: pooled @ fc_w + fc_b, log_softmax ----------------
__global__ void head_kernel(const float* __restrict__ psum, const float* __restrict__ pcnt,
                            const float* __restrict__ fcw, const float* __restrict__ fcb,
                            float* __restrict__ out, int G, int C2, int NCLS)
{
    int g = threadIdx.x;
    if (g >= G) return;
    float inv = 1.0f / fmaxf(pcnt[g], 1.0f);
    float logit[16];
    for (int k = 0; k < NCLS; ++k) logit[k] = fcb[k];
    for (int c = 0; c < C2; ++c) {
        float p = psum[(size_t)g * C2 + c] * inv;
        for (int k = 0; k < NCLS; ++k) logit[k] += p * fcw[(size_t)c * NCLS + k];
    }
    float mx = logit[0];
    for (int k = 1; k < NCLS; ++k) mx = fmaxf(mx, logit[k]);
    float se = 0.f;
    for (int k = 0; k < NCLS; ++k) se += expf(logit[k] - mx);
    float lse = mx + logf(se);
    for (int k = 0; k < NCLS; ++k) out[(size_t)g * NCLS + k] = logit[k] - lse;
}

// ---------------------------------------------------------------------------
extern "C" void kernel_launch(void* const* d_in, const int* in_sizes, int n_in,
                              void* d_out, int out_size, void* d_ws, size_t ws_size,
                              hipStream_t stream)
{
    const float* x    = (const float*)d_in[0];
    const int*   ei   = (const int*)  d_in[1];
    const int*   batch= (const int*)  d_in[2];
    const float* W1   = (const float*)d_in[3];
    const float* asw1 = (const float*)d_in[4];
    const float* adw1 = (const float*)d_in[5];
    const float* b1   = (const float*)d_in[6];
    const float* W2   = (const float*)d_in[7];
    const float* asw2 = (const float*)d_in[8];
    const float* adw2 = (const float*)d_in[9];
    const float* b2   = (const float*)d_in[10];
    const float* fcw  = (const float*)d_in[11];
    const float* fcb  = (const float*)d_in[12];
    float* out = (float*)d_out;

    const int F_IN = 256, H = 4, C1 = 16, C2 = 32, G = 64, NCLS = 10;
    const int F1 = H * C1;                       // 64
    const int N    = in_sizes[0] / F_IN;
    const int E    = in_sizes[1] / 2;
    const int Etot = E + N;

    // ---- workspace layout (floats) ----
    float* ws = (float*)d_ws;
    float* h1   = ws;                ws += (size_t)N * F1;      // [N,64]
    float* out1 = ws;                ws += (size_t)N * F1;      // [N,64] agg -> h1b
    float* h2   = ws;                ws += (size_t)N * C2;      // [N,32]
    float* out2 = ws;                ws += (size_t)N * C2;      // [N,32]
    float* as1  = ws;                ws += (size_t)N * H;
    float* ad1  = ws;                ws += (size_t)N * H;
    float* m1   = ws;                ws += (size_t)N * H;
    float* den1 = ws;                ws += (size_t)N * H;
    float* as2  = ws;                ws += (size_t)N;
    float* ad2  = ws;                ws += (size_t)N;
    float* m2   = ws;                ws += (size_t)N;
    float* den2 = ws;                ws += (size_t)N;
    float* sc1  = ws;                ws += (size_t)Etot * H;
    float* sc2  = ws;                ws += (size_t)Etot;
    float* psum = ws;                ws += (size_t)G * C2;
    float* pcnt = ws;                ws += (size_t)G;

    const float NEG_INF = -__builtin_inff();
    auto blk = [](size_t n) { return (unsigned)((n + 255) / 256); };

    // ---- init accumulators (every call: deterministic / graph-safe) ----
    fill_kernel<<<blk((size_t)N*F1), 256, 0, stream>>>(out1, (size_t)N*F1, 0.f);
    fill_kernel<<<blk((size_t)N*C2), 256, 0, stream>>>(out2, (size_t)N*C2, 0.f);
    fill_kernel<<<blk((size_t)N*H),  256, 0, stream>>>(den1, (size_t)N*H, 0.f);
    fill_kernel<<<blk((size_t)N*H),  256, 0, stream>>>(m1,   (size_t)N*H, NEG_INF);
    fill_kernel<<<blk((size_t)N),    256, 0, stream>>>(den2, (size_t)N, 0.f);
    fill_kernel<<<blk((size_t)N),    256, 0, stream>>>(m2,   (size_t)N, NEG_INF);
    fill_kernel<<<blk((size_t)G*C2 + G), 256, 0, stream>>>(psum, (size_t)G*C2 + G, 0.f);

    const unsigned gemmBlocks = (unsigned)((N / 16 + 7) / 8);

    // ================= layer 1 =================
    gemm_wmma_splitbf16<256, 64><<<gemmBlocks, 256, 0, stream>>>(x, W1, h1, N);
    att_coef_kernel<<<blk((size_t)N*H), 256, 0, stream>>>(h1, asw1, adw1, as1, ad1, N, H, C1);
    edge_max_kernel<<<blk(Etot), 256, 0, stream>>>(ei, E, Etot, as1, ad1, sc1, m1, H);
    edge_exp_kernel<<<blk(Etot), 256, 0, stream>>>(ei, E, Etot, sc1, m1, den1, H);
    edge_scatter_kernel<<<blk((size_t)Etot*F1), 256, 0, stream>>>(
        ei, E, Etot, sc1, den1, h1, out1, /*cshift=*/6, /*hshift=*/4, H);
    bias_elu_kernel<<<blk((size_t)N*F1), 256, 0, stream>>>(out1, b1, (size_t)N*F1, F1 - 1);

    // ================= layer 2 =================
    gemm_wmma_splitbf16<64, 32><<<gemmBlocks, 256, 0, stream>>>(out1, W2, h2, N);
    att_coef_kernel<<<blk((size_t)N), 256, 0, stream>>>(h2, asw2, adw2, as2, ad2, N, 1, C2);
    edge_max_kernel<<<blk(Etot), 256, 0, stream>>>(ei, E, Etot, as2, ad2, sc2, m2, 1);
    edge_exp_kernel<<<blk(Etot), 256, 0, stream>>>(ei, E, Etot, sc2, m2, den2, 1);
    edge_scatter_kernel<<<blk((size_t)Etot*C2), 256, 0, stream>>>(
        ei, E, Etot, sc2, den2, h2, out2, /*cshift=*/5, /*hshift=*/5, 1);
    bias_elu_kernel<<<blk((size_t)N*C2), 256, 0, stream>>>(out2, b2, (size_t)N*C2, C2 - 1);

    // ================= pool + classifier head =================
    pool_kernel<<<blk((size_t)N*C2), 256, 0, stream>>>(out2, batch, psum, pcnt, N, /*cshift=*/5);
    head_kernel<<<1, G, 0, stream>>>(psum, pcnt, fcw, fcb, out, G, C2, NCLS);
}
